// OML_16277926052524
// MI455X (gfx1250) — matrix operand order repaired and checked by reading, not compiled, not run on hardware
//
#include <hip/hip_runtime.h>

typedef __attribute__((ext_vector_type(16))) _Float16 v16h;
typedef __attribute__((ext_vector_type(8)))  float    v8f;
typedef __attribute__((ext_vector_type(2)))  float    v2f;
typedef __attribute__((ext_vector_type(4)))  unsigned int u32x4;
typedef __attribute__((ext_vector_type(8)))  int      i32x8;
typedef __attribute__((ext_vector_type(4)))  int      i32x4;

#if defined(__has_builtin)
# if __has_builtin(__builtin_amdgcn_wmma_f32_16x16x4_f32)
#  define HAVE_WMMA_F32X4 1
# endif
# if __has_builtin(__builtin_amdgcn_tensor_load_to_lds)
#  define HAVE_TDM 1
# endif
#endif

#define NB 384   // batch
#define ND 512   // embed dim
#define NL 40    // labels
#define BW_DWORDS (12 * 3 * 32 * 16 / 2)   // 9216 dwords = 36864 B packed B blob

// ---------------------------------------------------------------------------
// Kernel 0: zero the 40-float accumulator (graph-safe)
// ---------------------------------------------------------------------------
__global__ void k_zero(float* __restrict__ p, int n) {
  int i = blockIdx.x * blockDim.x + threadIdx.x;
  if (i < n) p[i] = 0.f;
}

// ---------------------------------------------------------------------------
// Kernel 1: row-normalize embeddings (F.normalize, eps=1e-12)
// ---------------------------------------------------------------------------
__global__ __launch_bounds__(128) void k_normalize(const float* __restrict__ e,
                                                   float* __restrict__ x) {
  int row = blockIdx.x;
  int tid = threadIdx.x;
  const float* er = e + (size_t)row * ND;
  float s = 0.f;
  for (int i = tid; i < ND; i += 128) { float v = er[i]; s += v * v; }
  __shared__ float red[128];
  red[tid] = s;
  __syncthreads();
  for (int off = 64; off > 0; off >>= 1) {
    if (tid < off) red[tid] += red[tid + off];
    __syncthreads();
  }
  float inv = 1.f / fmaxf(sqrtf(red[0]), 1e-12f);
  float* xr = x + (size_t)row * ND;
  for (int i = tid; i < ND; i += 128) xr[i] = er[i] * inv;
}

// ---------------------------------------------------------------------------
// Kernel 2: sim = x @ x^T via V_WMMA_F32_16X16X4_F32 (full f32 precision).
// ---------------------------------------------------------------------------
__global__ __launch_bounds__(32) void k_sim(const float* __restrict__ x,
                                            float* __restrict__ sim) {
  int rt = blockIdx.y, ct = blockIdx.x;
  int lane = threadIdx.x;
  int m16 = lane & 15, hi = lane >> 4;
  const float* arow = x + (size_t)(rt * 16 + m16) * ND;
  const float* brow = x + (size_t)(ct * 16 + m16) * ND;
#if HAVE_WMMA_F32X4
  v8f c = {};
  for (int k = 0; k < ND; k += 4) {
    int kk = k + hi * 2;
    v2f a = { arow[kk], arow[kk + 1] };
    v2f b = { brow[kk], brow[kk + 1] };
    c = __builtin_amdgcn_wmma_f32_16x16x4_f32(false, a, false, b,
                                              (short)0, c, false, false);
  }
#pragma unroll
  for (int i = 0; i < 8; ++i)
    sim[(size_t)(rt * 16 + i + hi * 8) * NB + ct * 16 + m16] = c[i];
#else
#pragma unroll
  for (int i = 0; i < 8; ++i) {
    const float* r = x + (size_t)(rt * 16 + i + hi * 8) * ND;
    float acc = 0.f;
    for (int k = 0; k < ND; ++k) acc += r[k] * brow[k];
    sim[(size_t)(rt * 16 + i + hi * 8) * NB + ct * 16 + m16] = acc;
  }
#endif
}

// ---------------------------------------------------------------------------
// Kernel 3: pre-pack B tiles (labels + ones column, f16) in WMMA B-layout.
// bw[kk(12)][t(3)][lane(32)][16 halves]
// ---------------------------------------------------------------------------
__global__ __launch_bounds__(32) void k_pack_b(const int* __restrict__ labels,
                                               _Float16* __restrict__ bw) {
  int kk = blockIdx.x, t = blockIdx.y;
  int lane = threadIdx.x;
  int m16 = lane & 15, hi = lane >> 4;
  int n = t * 16 + m16;
  _Float16* dst = bw + ((size_t)(kk * 3 + t) * 32 + lane) * 16;
#pragma unroll
  for (int h = 0; h < 16; ++h) {
    int K = (h < 8 ? h : h + 8) + (hi ? 8 : 0);
    int k = kk * 32 + K;
    float v = 0.f;
    if (n < NL)       v = (float)labels[k * NL + n];
    else if (n == NL) v = 1.f;      // ones column -> rk_all
    dst[h] = (_Float16)v;
  }
}

// ---------------------------------------------------------------------------
// Kernel 4: fused B^3 kernel. Packed-B blob staged into LDS once per block
// via the Tensor Data Mover; sg 16x32 f16 A-tile built in registers
// (fma + med3 + TRANS exp2/rcp); 3x v_wmma_f32_16x16x32_f16 with reuse_a.
// ---------------------------------------------------------------------------
__global__ __launch_bounds__(128) void k_pairs(const float* __restrict__ sim,
                                               const _Float16* __restrict__ bw,
                                               const int* __restrict__ labels,
                                               float* __restrict__ s_accum) {
  __shared__ __attribute__((aligned(32))) _Float16 bwl[12 * 3 * 32 * 16]; // 36864 B
  __shared__ float simq[NB];
  __shared__ float rkall_sh[4][16];
  __shared__ float s_local[48];

  int q = blockIdx.y;
  int tid = threadIdx.x;
  int wave = tid >> 5, lane = tid & 31;

  __builtin_prefetch(sim + (size_t)q * NB, 0, 3);
  __builtin_prefetch(labels + q * NL, 0, 3);

  for (int i = tid; i < NB; i += 128) simq[i] = sim[(size_t)q * NB + i];
  if (tid < 48) s_local[tid] = 0.f;

#if HAVE_TDM
  if (wave == 0) {
    // D# group 0/1 per CDNA5 ISA: 1D tile, 4B elems.
    unsigned lds_off = (unsigned)(unsigned long long)(&bwl[0]);
    unsigned long long ga = (unsigned long long)(const void*)bw;
    u32x4 g0 = { 0u, 0u, 0u, 0u };
    g0[0] = 1u;                                        // count=1 (user D#)
    g0[1] = lds_off;                                   // lds_addr
    g0[2] = (unsigned)(ga & 0xFFFFFFFFu);              // global_addr[31:0]
    g0[3] = (unsigned)((ga >> 32) & 0x01FFFFFFu) | (2u << 30); // addr[56:32]|type=2
    i32x8 g1 = { 0, 0, 0, 0, 0, 0, 0, 0 };
    const int td0 = BW_DWORDS;                         // 9216 dword elements
    g1[0] = (2 << 16);                                 // data_size=2 -> 4 bytes
    g1[1] = (td0 & 0xFFFF) << 16;                      // tensor_dim0[15:0]
    g1[2] = ((td0 >> 16) & 0xFFFF) | (1 << 16);        // tensor_dim0[31:16], tensor_dim1=1
    g1[3] = (td0 & 0xFFFF) << 16;                      // tile_dim0 = 9216
    g1[4] = 0;                                         // tile_dim1/2 unused
    g1[5] = td0;                                       // tensor_dim0_stride[31:0]
    i32x4 g2 = { 0, 0, 0, 0 };
    i32x4 g3 = { 0, 0, 0, 0 };
# if __clang_major__ >= 23
    i32x8 g4 = { 0, 0, 0, 0, 0, 0, 0, 0 };
    __builtin_amdgcn_tensor_load_to_lds(g0, g1, g2, g3, g4, 0);
# else
    __builtin_amdgcn_tensor_load_to_lds(g0, g1, g2, g3, 0);
# endif
    __builtin_amdgcn_s_wait_tensorcnt((short)0);
  }
#else
  for (int i = tid; i < BW_DWORDS; i += 128)
    ((unsigned*)bwl)[i] = ((const unsigned*)bw)[i];
#endif
  __syncthreads();

  int m16 = lane & 15, hi = lane >> 4;
  int jbase = (blockIdx.x * 4 + wave) * 16;

  // u = clip(-(sim[q,k]-sim[q,j])/T) * log2(e)  ==  clamp(sjC - simqk*C, +-CLIP)
  const float C    = 144.26950408889634f;   // 100 * log2(e)
  const float CLIP = 72.134752044448170f;   // 50  * log2(e)
  float sjC = simq[jbase + m16] * C;

  const v16h* bwv = (const v16h*)bwl;
  v8f cacc[3] = { v8f{}, v8f{}, v8f{} };

  for (int kk = 0; kk < 12; ++kk) {
    int k0 = kk * 32;
    v16h a;
#pragma unroll
    for (int h = 0; h < 16; ++h) {
      // 16-bit A 16x32 layout: h<8 -> K=h(+8*hi), h>=8 -> K=h+8(+8*hi)
      int K = (h < 8 ? h : h + 8) + (hi ? 8 : 0);
      float u  = fmaf(simq[k0 + K], -C, sjC);             // (sim_j - sim_k)*100*log2e
      u        = fminf(fmaxf(u, -CLIP), CLIP);            // v_med3_f32
      float ex = __builtin_amdgcn_exp2f(u);               // TRANS
      float sg = __builtin_amdgcn_rcpf(1.f + ex);         // TRANS
      a[h] = (_Float16)sg;
    }
    v16h b0 = bwv[(kk * 3 + 0) * 32 + lane];
    v16h b1 = bwv[(kk * 3 + 1) * 32 + lane];
    v16h b2 = bwv[(kk * 3 + 2) * 32 + lane];
    cacc[0] = __builtin_amdgcn_wmma_f32_16x16x32_f16(false, a, false, b0,
                                                     (short)0, cacc[0], true,  false);
    cacc[1] = __builtin_amdgcn_wmma_f32_16x16x32_f16(false, a, false, b1,
                                                     (short)0, cacc[1], true,  false);
    cacc[2] = __builtin_amdgcn_wmma_f32_16x16x32_f16(false, a, false, b2,
                                                     (short)0, cacc[2], false, false);
  }

  // rk_all sums live in column n=40 (tile 2, local col 8): lanes 8 and 24.
  if (m16 == 8) {
#pragma unroll
    for (int i = 0; i < 8; ++i) rkall_sh[wave][hi * 8 + i] = cacc[2][i];
  }
  __syncthreads();

  // C layout: VGPR i = row (i + 8*hi) of the j-tile, col = lane%16.
#pragma unroll
  for (int t = 0; t < 3; ++t) {
    int n = t * 16 + m16;
    if (n < NL) {
      float posq = (float)labels[q * NL + n];
      if (posq != 0.f) {
        float acc = 0.f;
#pragma unroll
        for (int i = 0; i < 8; ++i) {
          int m = hi * 8 + i;
          float posj = (float)labels[(jbase + m) * NL + n];
          if (posj != 0.f) {
            float rkpos = 1.f + cacc[t][i] - 0.5f * posj;  // 1 + S_l - 0.5*pos[j]
            float rkall = 0.5f + rkall_sh[wave][m];        // 1 + S_all - 0.5
            acc += posq * posj * rkpos * __builtin_amdgcn_rcpf(rkall);
          }
        }
        if (acc != 0.f) atomicAdd(&s_local[n], acc);       // ds_add_f32
      }
    }
  }
  __syncthreads();
  if (tid < NL) atomicAdd(&s_accum[tid], s_local[tid]);
}

// ---------------------------------------------------------------------------
// Kernel 5: finalize -> scalar loss
// ---------------------------------------------------------------------------
__global__ __launch_bounds__(64) void k_finalize(const int* __restrict__ labels,
                                                 const float* __restrict__ s_accum,
                                                 float* __restrict__ out) {
  __shared__ float red_t[64];
  __shared__ float red_v[64];
  int tid = threadIdx.x;
  float term = 0.f, vld = 0.f;
  if (tid < NL) {
    float cnt = 0.f;
    for (int qq = 0; qq < NB; ++qq) cnt += (float)labels[qq * NL + tid];
    if (cnt > 1.f) {
      float ap = s_accum[tid] / fmaxf(cnt, 1.f) / (float)NB;
      term = 1.f - ap;
      vld = 1.f;
    }
  }
  red_t[tid] = term; red_v[tid] = vld;
  __syncthreads();
  for (int off = 32; off > 0; off >>= 1) {
    if (tid < off) { red_t[tid] += red_t[tid + off]; red_v[tid] += red_v[tid + off]; }
    __syncthreads();
  }
  if (tid == 0) out[0] = red_t[0] / fmaxf(red_v[0], 1.f);
}

// ---------------------------------------------------------------------------
extern "C" void kernel_launch(void* const* d_in, const int* in_sizes, int n_in,
                              void* d_out, int out_size, void* d_ws, size_t ws_size,
                              hipStream_t stream) {
  const float* emb   = (const float*)d_in[0];  // [384,512] f32
  const int*  labels = (const int*)d_in[1];    // [384,40] i32

  char* ws = (char*)d_ws;
  float*    x     = (float*)(ws);                                   // 786432 B
  float*    sim   = (float*)(ws + 786432);                          // 589824 B
  _Float16* bw    = (_Float16*)(ws + 786432 + 589824);              //  36864 B
  float*    s_acc = (float*)(ws + 786432 + 589824 + 36864);         //    160 B
  float*    out   = (float*)d_out;

  k_zero     <<<1, 64, 0, stream>>>(s_acc, NL);
  k_normalize<<<NB, 128, 0, stream>>>(emb, x);
  k_sim      <<<dim3(NB / 16, NB / 16), 32, 0, stream>>>(x, sim);
  k_pack_b   <<<dim3(12, 3), 32, 0, stream>>>(labels, bw);
  k_pairs    <<<dim3(NB / 64, NB), 128, 0, stream>>>(sim, bw, labels, s_acc);
  k_finalize <<<1, 64, 0, stream>>>(labels, s_acc, out);
}
